// MDN_38946763440904
// MI455X (gfx1250) — compile-verified, matrix operand order checked by reading, not run on hardware
//
#include <hip/hip_runtime.h>
#include <hip/hip_bf16.h>

typedef __attribute__((ext_vector_type(16))) _Float16 v16h;
typedef __attribute__((ext_vector_type(8)))  float    v8f;

#define HW      (512 * 512)
#define NBAT    4
#define NGROUPS ((NBAT * HW) / 16)   // 65536 groups of 16 pixels

struct MdnParams {
    const float* x;
    const float* w1; const float* b1;
    const float* w2; const float* b2;
    const float* w3; const float* b3;
    const float* hw[9];   // mu,sg,pi  for r,g,b
    const float* hb[9];
    float* out;
};

__device__ __forceinline__ v8f wmma_f16(v16h a, v16h b, v8f c) {
    // D = A(16x32 f16) * B(32x16 f16) + C(16x16 f32)
    return __builtin_amdgcn_wmma_f32_16x16x32_f16(
        /*neg_a=*/false, a, /*neg_b=*/false, b,
        /*c_mod=*/(short)0, c, /*reuse_a=*/false, /*reuse_b=*/false);
}

// ReLU in one VALU op: med3(x, 0, +inf) clamps to [0, inf).
__device__ __forceinline__ float relu1(float x) {
    return __builtin_amdgcn_fmed3f(x, 0.0f, __builtin_inff());
}

// Build one 16x32 f16 A-operand tile from row-major weights w[m*ld + k].
// Layout (ISA 7.12.2, 16-bit A 16x32): lanes 0-15 row M=l16 (K 0-7 / 16-23),
// lanes 16-31 row M=l16 (K 8-15 / 24-31).
__device__ __forceinline__ v16h loadA(const float* w, int ld, int kmax,
                                      int mOff, int hi, int l16) {
    const int m = mOff + l16;
    v16h a;
#pragma unroll
    for (int j = 0; j < 16; ++j) {
        const int k = (j < 8 ? j : j + 8) + hi * 8;
        a[j] = (_Float16)((k < kmax) ? w[m * ld + k] : 0.0f);
    }
    return a;
}

// Layer-1 A tile: 5 real input channels in K 0-4, bias in K slot 5
// (paired with a constant-1.0 activation in B).  Only lanes 0-15 carry
// K 0-7, so the whole augmented column lives in the low half.
__device__ __forceinline__ v16h loadA1(const float* w, const float* b,
                                       int mOff, int hi, int l16) {
    const int m = mOff + l16;
    v16h a;
#pragma unroll
    for (int j = 0; j < 16; ++j) {
        const int k = (j < 8 ? j : j + 8) + hi * 8;
        float v = 0.0f;
        if (k < 5)       v = w[m * 5 + k];
        else if (k == 5) v = b[m];
        a[j] = (_Float16)v;
    }
    return a;
}

// Bias as a C-operand (per-lane constant across the N dimension).
__device__ __forceinline__ v8f loadBias(const float* b, int mOff, int hi) {
    v8f c;
#pragma unroll
    for (int r = 0; r < 8; ++r) c[r] = b[mOff + hi * 8 + r];
    return c;
}

// D-layout of this layer == B-layout of the next layer, lane-locally:
// B[j] = relu(D_tile0[j]), B[j+8] = relu(D_tile1[j]).  No cross-lane moves.
__device__ __forceinline__ v16h relu_pack(v8f d0, v8f d1) {
    v16h b;
#pragma unroll
    for (int j = 0; j < 8; ++j) {
        b[j]     = (_Float16)relu1(d0[j]);
        b[j + 8] = (_Float16)relu1(d1[j]);
    }
    return b;
}

__global__ __launch_bounds__(256) void mdn_kernel(MdnParams p) {
    const int lane = (int)(threadIdx.x & 31u);
    const int hi   = (lane >> 4) & 1;
    const int l16  = lane & 15;

    // ---- weights + biases resident in VGPRs for the whole kernel ----
    const v16h A1a = loadA1(p.w1, p.b1, 0, hi, l16);   // bias folded into K=5
    const v16h A1b = loadA1(p.w1, p.b1, 16, hi, l16);
    const v16h A2a = loadA(p.w2, 32, 32, 0, hi, l16);
    const v16h A2b = loadA(p.w2, 32, 32, 16, hi, l16);
    const v16h A3a = loadA(p.w3, 32, 32, 0, hi, l16);
    const v16h A3b = loadA(p.w3, 32, 32, 16, hi, l16);
    v16h AH[9];
    v8f  CH[9];
#pragma unroll
    for (int t = 0; t < 9; ++t) {
        AH[t] = loadA(p.hw[t], 32, 32, 0, hi, l16);   // 16x32 head weights
        CH[t] = loadBias(p.hb[t], 0, hi);
    }
    const v8f C2a = loadBias(p.b2, 0, hi), C2b = loadBias(p.b2, 16, hi);
    const v8f C3a = loadBias(p.b3, 0, hi), C3b = loadBias(p.b3, 16, hi);
    const v8f Z0  = {0.f, 0.f, 0.f, 0.f, 0.f, 0.f, 0.f, 0.f};

    const int wave  = (int)((blockIdx.x * blockDim.x + threadIdx.x) >> 5);
    const int waves = (int)((gridDim.x * blockDim.x) >> 5);

    for (int g = wave; g < NGROUPS; g += waves) {
        // 16 contiguous pixels; lanes L and L+16 handle the same pixel
        // (upper/lower 8 output rows), so loads are shared-pattern.
        const int gp = g * 16 + l16;
        const int b  = gp >> 18;          // / HW  (HW = 2^18)
        const int s  = gp & (HW - 1);
        const float* xb = p.x + (size_t)b * 5 * HW + s;
        const float x0 = xb[0];
        const float x1 = xb[(size_t)1 * HW];
        const float x2 = xb[(size_t)2 * HW];
        const float x3 = xb[(size_t)3 * HW];
        const float x4 = xb[(size_t)4 * HW];

        // ---- layer 1: K=5 (+1 bias channel) zero-padded to 32.
        // Branchless masking keeps EXEC all-ones around the WMMAs.
        const float msk = hi ? 0.0f : 1.0f;   // K 0-7 slots live in lanes 0-15
        v16h B;
#pragma unroll
        for (int j = 0; j < 16; ++j) B[j] = (_Float16)0.0f;
        B[0] = (_Float16)(x0 * msk);
        B[1] = (_Float16)(x1 * msk);
        B[2] = (_Float16)(x2 * msk);
        B[3] = (_Float16)(x3 * msk);
        B[4] = (_Float16)(x4 * msk);
        B[5] = (_Float16)msk;                 // constant-1 bias channel

        v8f d0 = wmma_f16(A1a, B, Z0);
        v8f d1 = wmma_f16(A1b, B, Z0);
        v16h act = relu_pack(d0, d1);

        d0 = wmma_f16(A2a, act, C2a);
        d1 = wmma_f16(A2b, act, C2b);
        act = relu_pack(d0, d1);

        d0 = wmma_f16(A3a, act, C3a);
        d1 = wmma_f16(A3b, act, C3b);
        const v16h lat = relu_pack(d0, d1);

        const float xres0 = x0, xres1 = x1, xres2 = x2;
#pragma unroll
        for (int c = 0; c < 3; ++c) {
            const float xres = (c == 0) ? xres0 : (c == 1 ? xres1 : xres2);
            v8f Dm = wmma_f16(AH[3 * c + 0], lat, CH[3 * c + 0]);  // mu + bias
            v8f Ds = wmma_f16(AH[3 * c + 1], lat, CH[3 * c + 1]);  // pre-softplus
            v8f Dp = wmma_f16(AH[3 * c + 2], lat, CH[3 * c + 2]);  // pre-softmax

            // softmax over 16 mixture components: 8 local + 8 in lane^16
            float zmax = Dp[0];
#pragma unroll
            for (int r = 1; r < 8; ++r) zmax = fmaxf(zmax, Dp[r]);
            zmax = fmaxf(zmax, __shfl_xor(zmax, 16, 32));
            float e[8];
            float esum = 0.0f;
#pragma unroll
            for (int r = 0; r < 8; ++r) { e[r] = __expf(Dp[r] - zmax); esum += e[r]; }
            esum += __shfl_xor(esum, 16, 32);
            const float rinv = __builtin_amdgcn_rcpf(esum);

            // out tensor t=[3c+h], shape [4,16,HW]; this lane owns rows hi*8..hi*8+7
            float* omu = p.out + (((size_t)(3 * c + 0) * NBAT + b) * 16 + hi * 8) * (size_t)HW + s;
            float* osg = p.out + (((size_t)(3 * c + 1) * NBAT + b) * 16 + hi * 8) * (size_t)HW + s;
            float* opi = p.out + (((size_t)(3 * c + 2) * NBAT + b) * 16 + hi * 8) * (size_t)HW + s;
#pragma unroll
            for (int r = 0; r < 8; ++r) {
                __builtin_nontemporal_store(Dm[r] + xres, omu + (size_t)r * HW);
                const float z  = Ds[r];
                const float sp = (z > 15.0f) ? z : __logf(1.0f + __expf(z));
                __builtin_nontemporal_store(sp, osg + (size_t)r * HW);
                __builtin_nontemporal_store(e[r] * rinv, opi + (size_t)r * HW);
            }
        }
    }
}

extern "C" void kernel_launch(void* const* d_in, const int* in_sizes, int n_in,
                              void* d_out, int out_size, void* d_ws, size_t ws_size,
                              hipStream_t stream) {
    MdnParams p;
    p.x  = (const float*)d_in[0];
    p.w1 = (const float*)d_in[1]; p.b1 = (const float*)d_in[2];
    p.w2 = (const float*)d_in[3]; p.b2 = (const float*)d_in[4];
    p.w3 = (const float*)d_in[5]; p.b3 = (const float*)d_in[6];
    for (int t = 0; t < 9; ++t) {           // rmu,rsg,rpi,gmu,gsg,gpi,bmu,bsg,bpi
        p.hw[t] = (const float*)d_in[7 + 2 * t];
        p.hb[t] = (const float*)d_in[8 + 2 * t];
    }
    p.out = (float*)d_out;

    // 1024 blocks x 8 waves = 8192 waves; 65536 groups -> 8 groups (128 px) per wave.
    hipLaunchKernelGGL(mdn_kernel, dim3(1024), dim3(256), 0, stream, p);
}
